// LSTMFeatureExtractor_7464653160605
// MI455X (gfx1250) — compile-verified, matrix operand order checked by reading, not run on hardware
//
#include <hip/hip_runtime.h>
#include <cstdint>
#include <cstddef>

typedef __attribute__((ext_vector_type(16))) _Float16 v16h;
typedef __attribute__((ext_vector_type(8)))  _Float16 v8h;
typedef __attribute__((ext_vector_type(8)))  float    v8f;
typedef __attribute__((ext_vector_type(4)))  uint32_t v4u;
typedef __attribute__((ext_vector_type(8)))  int      v8i;
typedef __attribute__((ext_vector_type(4)))  int      v4i;

namespace {
constexpr int B = 64, T = 512, DIN = 862, H = 1024, G = 4096;
constexpr int BT   = B * T;
constexpr int DINP = 896;            // DIN padded up to multiple of 32 (28 K-chunks)
constexpr int WSTR = H + 8;          // LDS row stride (f16) produced by TDM pad injection
constexpr int REC_BLOCKS = H / 16;   // 64 persistent workgroups, one per 16 hidden units
constexpr size_t SMEM_REC =
    (size_t)64 * WSTR * sizeof(_Float16) +   // W_hh slice: 4 chunks x 16 rows x (1024+8)
    (size_t)4 * 64 * 16 * sizeof(float);     // f32 gate staging buffer [4][B][16]
}

// ---------------- WMMA helpers (CDNA5 16x16x32 f16 -> f32) ----------------

static __device__ __forceinline__ v8f wmma16(v16h a, v16h b, v8f c) {
  return __builtin_amdgcn_wmma_f32_16x16x32_f16(false, a, false, b, (short)0, c,
                                                false, false);
}

// 16x32 operand fragment from row-major [rows x stride] f16 (ISA 7.12.2 layout):
// lanes 0-15 rows 0-15 with K 0..7/16..23, lanes 16-31 same rows K 8..15/24..31.
static __device__ __forceinline__ v16h load_frag(const _Float16* base, int stride,
                                                 int lane) {
  const int r  = lane & 15;
  const int kh = (lane & 16) ? 8 : 0;
  const _Float16* p = base + r * stride + kh;
  v8h lo = *(const v8h*)(p);
  v8h hi = *(const v8h*)(p + 16);
  return __builtin_shufflevector(lo, hi, 0, 1, 2, 3, 4, 5, 6, 7,
                                 8, 9, 10, 11, 12, 13, 14, 15);
}

// ---------------- Tensor Data Mover: 2D tile global -> LDS ----------------
// D# per CDNA5 ISA ch.8: group0 = {count, lds_addr, global_addr, type=2},
// group1 = {data_size=8B, pad ctl, tensor/tile dims, dim0 stride}.
// Tile rows are tile_d0 8-byte elements, tile_d1 rows, row stride d0_stride
// elements; LDS gets pad_amt+1 DWORDs injected every 2^(pad_int+1) DWORDs.
static __device__ __forceinline__ uint32_t lds_off(const void* p) {
  return (uint32_t)(uintptr_t)p;     // generic LDS pointer: addr[31:0] = LDS offset
}

static __device__ __forceinline__ void tdm_load_2d(uint32_t ldsOff, const void* gptr,
                                                   uint32_t tile_d0, uint32_t tile_d1,
                                                   uint64_t d0_stride,
                                                   uint32_t pad_int, uint32_t pad_amt) {
  const uint64_t ga = (uint64_t)gptr;
  uint64_t a0 = 1ull | ((uint64_t)ldsOff << 32);                  // count=1, lds_addr
  uint64_t a1 = (ga & ((1ull << 57) - 1)) | (2ull << 62);         // global_addr, type=2
  uint64_t b0 = ((uint64_t)3 << 16)                               // data_size = 8B
              | (1ull << 20)                                      // pad_enable
              | ((uint64_t)pad_int << 22) | ((uint64_t)pad_amt << 25)
              | ((uint64_t)tile_d0 << 48);                        // tensor_dim0[79:48]
  uint64_t b1 = ((uint64_t)tile_d0 >> 16)
              | ((uint64_t)tile_d1 << 16)                         // tensor_dim1[111:80]
              | ((uint64_t)tile_d0 << 48);                        // tile_dim0[127:112]
  uint64_t b2 = (uint64_t)tile_d1                                 // tile_dim1[143:128]
              | (d0_stride << 32);                                // dim0_stride[207:160]
  uint64_t b3 = d0_stride >> 32;
  v4u g0 = {(uint32_t)a0, (uint32_t)(a0 >> 32), (uint32_t)a1, (uint32_t)(a1 >> 32)};
  v8i g1 = {(int)b0, (int)(b0 >> 32), (int)b1, (int)(b1 >> 32),
            (int)b2, (int)(b2 >> 32), (int)b3, (int)(b3 >> 32)};
  v4i z4 = {0, 0, 0, 0};
#if defined(__clang_major__) && __clang_major__ >= 23
  v8i z8 = {0, 0, 0, 0, 0, 0, 0, 0};
  __builtin_amdgcn_tensor_load_to_lds(g0, g1, z4, z4, z8, 0);
#else
  __builtin_amdgcn_tensor_load_to_lds(g0, g1, z4, z4, 0);
#endif
}

// ---------------- utility kernels ----------------

__global__ void zero_u32_kernel(uint32_t* __restrict__ p, long n) {
  for (long i = (long)blockIdx.x * blockDim.x + threadIdx.x; i < n;
       i += (long)gridDim.x * blockDim.x)
    p[i] = 0u;
}

__global__ void cvt_pad_f16_kernel(const float* __restrict__ src,
                                   _Float16* __restrict__ dst, int rows, int kin,
                                   int kout) {
  long i = (long)blockIdx.x * blockDim.x + threadIdx.x;
  long total = (long)rows * kout;
  if (i >= total) return;
  int r = (int)(i / kout);
  int k = (int)(i % kout);
  dst[i] = (k < kin) ? (_Float16)src[(long)r * kin + k] : (_Float16)0.f;
}

__global__ void bias_sum_kernel(const float* __restrict__ a,
                                const float* __restrict__ b,
                                float* __restrict__ o, int n) {
  int i = blockIdx.x * blockDim.x + threadIdx.x;
  if (i < n) o[i] = a[i] + b[i];
}

// ---------------- big input-projection GEMM ----------------
// C[M,N] (f16) = A[M,K] * Bm[N,K]^T + bias[N].  TDM double-buffered: wave 0
// DMAs chunk k+1 (64 rows x 64B, +16B LDS pad -> 80B row) while all 8 waves
// run WMMA on chunk k.  s_wait_tensorcnt(2) retires chunk k (in-order).
__global__ void __launch_bounds__(256)
wmma_gemm_bias_kernel(const _Float16* __restrict__ A, const _Float16* __restrict__ Bm,
                      const float* __restrict__ bias, _Float16* __restrict__ C,
                      int M, int N, int K) {
  __shared__ __align__(16) _Float16 As[2][64 * 40];
  __shared__ __align__(16) _Float16 Bs[2][64 * 40];

  const int tid  = threadIdx.x;
  const int lane = tid & 31;
  const int wave = tid >> 5;
  const int n0 = blockIdx.x * 64;
  const int m0 = blockIdx.y * 64;
  const int mt  = wave >> 1;          // M-tile 0..3
  const int ntp = (wave & 1) * 2;     // N-tile pair base: 0 or 2
  const int nk = K / 32;
  const uint64_t strideE = (uint64_t)K / 4;   // row stride in 8B elements

  v8f acc0 = {};
  v8f acc1 = {};

  if (wave == 0) {                    // prologue: chunk 0 in flight
    tdm_load_2d(lds_off(As[0]), A + (long)m0 * K, 8, 64, strideE, 3, 3);
    tdm_load_2d(lds_off(Bs[0]), Bm + (long)n0 * K, 8, 64, strideE, 3, 3);
  }

  for (int kc = 0; kc < nk; ++kc) {
    const int p = kc & 1;
    __syncthreads();                  // everyone done reading buf p^1
    if (wave == 0) {
      if (kc + 1 < nk) {
        tdm_load_2d(lds_off(As[p ^ 1]), A + (long)m0 * K + (kc + 1) * 32, 8, 64,
                    strideE, 3, 3);
        tdm_load_2d(lds_off(Bs[p ^ 1]), Bm + (long)n0 * K + (kc + 1) * 32, 8, 64,
                    strideE, 3, 3);
        __builtin_amdgcn_s_wait_tensorcnt((short)2);   // chunk kc landed
      } else {
        __builtin_amdgcn_s_wait_tensorcnt((short)0);
      }
    }
    __syncthreads();                  // chunk kc visible to all waves

    v16h a  = load_frag(As[p] + mt * 16 * 40, 40, lane);
    v16h b0 = load_frag(Bs[p] + ntp * 16 * 40, 40, lane);
    v16h b1 = load_frag(Bs[p] + (ntp + 1) * 16 * 40, 40, lane);
    acc0 = wmma16(a, b0, acc0);
    acc1 = wmma16(a, b1, acc1);
  }

  const int n  = lane & 15;
  const int rb = (lane >> 4) * 8;
  const int c0 = n0 + ntp * 16 + n;
  const int c1 = n0 + (ntp + 1) * 16 + n;
  const float bi0 = bias[c0];
  const float bi1 = bias[c1];
#pragma unroll
  for (int r = 0; r < 8; ++r) {
    const long m = (long)m0 + mt * 16 + rb + r;
    C[m * N + c0] = (_Float16)(acc0[r] + bi0);
    C[m * N + c1] = (_Float16)(acc1[r] + bi1);
  }
}

// ---------------- persistent recurrent kernel ----------------
// 64 blocks; block b owns hidden units u0=16*b and gate rows {q*H+u0..+15}.
// W_hh slice (128 KB + pad) DMA'd once into LDS via 4 TDM descriptors with
// 8B pad per 1024B -> 2064B row stride (bank-conflict-free fragment reads,
// K>=512 elements shifted +4).  Cell state in registers; h exchanged through
// a global f16 double buffer with a release/acquire grid barrier per step.
__global__ void __launch_bounds__(256)
lstm_rec_kernel(const _Float16* __restrict__ xg,   // [BT, G]
                const _Float16* __restrict__ Whh,  // [G, H] f16
                _Float16* __restrict__ hbuf,       // [2, B, H] f16 (parity 0 zeroed)
                int* __restrict__ bar,             // [T] arrival counters (zeroed)
                _Float16* __restrict__ out16,      // [BT, H] f16 or null
                float* __restrict__ out32) {       // [BT, H] f32 or null
  extern __shared__ __align__(16) char smem[];
  _Float16* Ws  = (_Float16*)smem;                                        // 4 x [16][WSTR]
  float*    gbf = (float*)(smem + (size_t)64 * WSTR * sizeof(_Float16));  // [4][64][16]

  const int tid  = threadIdx.x;
  const int lane = tid & 31;
  const int wave = tid >> 5;
  const int u0 = blockIdx.x * 16;

  if (wave == 0) {
    // one descriptor per gate chunk: 16 rows x 2048B, row stride H f16,
    // pad 2 DWORDs every 256 DWORDs -> LDS row stride 2064B (= WSTR f16)
#pragma unroll
    for (int q = 0; q < 4; ++q)
      tdm_load_2d(lds_off(Ws) + (uint32_t)q * (16 * WSTR * 2),
                  Whh + (size_t)(q * H + u0) * H, 256, 16, 256, 7, 1);
    __builtin_amdgcn_s_wait_tensorcnt((short)0);
  }
  __syncthreads();

  const int q   = wave >> 1;           // gate chunk handled by this wave
  const int mt0 = (wave & 1) * 2;      // batch tiles mt0, mt0+1

  float c_state[4];
#pragma unroll
  for (int e = 0; e < 4; ++e) c_state[e] = 0.f;

  for (int t = 0; t < T; ++t) {
    const int p = t & 1;
    if (t > 0) {
      if (tid == 0) {
        while (__hip_atomic_load(bar + (t - 1), __ATOMIC_ACQUIRE,
                                 __HIP_MEMORY_SCOPE_AGENT) < (int)gridDim.x)
          __builtin_amdgcn_s_sleep(1);
      }
      __syncthreads();
    }

    if (t + 1 < T) {   // global_prefetch_b8 on next step's xg slice
      const int b = tid >> 4, u = tid & 15;
      __builtin_prefetch(xg + ((long)b * T + (t + 1)) * G + u0 + u, 0, 1);
    }

    const _Float16* hprev = hbuf + (size_t)p * (B * H);

    v8f acc0 = {};
    v8f acc1 = {};
#pragma unroll 4
    for (int kc = 0; kc < H / 32; ++kc) {
      const int col = kc * 32 + ((kc & 16) ? 4 : 0);   // TDM mid-stream pad shift
      v16h bf = load_frag(Ws + (size_t)(q * 16) * WSTR + col, WSTR, lane);
      v16h a0 = load_frag(hprev + (size_t)(mt0 * 16) * H + kc * 32, H, lane);
      v16h a1 = load_frag(hprev + (size_t)((mt0 + 1) * 16) * H + kc * 32, H, lane);
      acc0 = wmma16(a0, bf, acc0);
      acc1 = wmma16(a1, bf, acc1);
    }

    {
      const int n  = lane & 15;
      const int rb = (lane >> 4) * 8;
#pragma unroll
      for (int r = 0; r < 8; ++r) {
        gbf[(q * 64 + mt0 * 16 + rb + r) * 16 + n]       = acc0[r];
        gbf[(q * 64 + (mt0 + 1) * 16 + rb + r) * 16 + n] = acc1[r];
      }
    }
    __syncthreads();

#pragma unroll
    for (int e = 0; e < 4; ++e) {
      const int idx = tid + e * 256;
      const int b = idx >> 4;
      const int u = idx & 15;
      const long row = (long)b * T + t;
      const long xbase = row * G + u0 + u;
      const float gi = gbf[(0 * 64 + b) * 16 + u] + (float)xg[xbase + 0 * H];
      const float gf = gbf[(1 * 64 + b) * 16 + u] + (float)xg[xbase + 1 * H];
      const float gg = gbf[(2 * 64 + b) * 16 + u] + (float)xg[xbase + 2 * H];
      const float go = gbf[(3 * 64 + b) * 16 + u] + (float)xg[xbase + 3 * H];
      const float i_ = 1.f / (1.f + __expf(-gi));
      const float f_ = 1.f / (1.f + __expf(-gf));
      const float g_ = tanhf(gg);
      const float o_ = 1.f / (1.f + __expf(-go));
      const float c = f_ * c_state[e] + i_ * g_;
      c_state[e] = c;
      const float h = o_ * tanhf(c);
      hbuf[(size_t)(p ^ 1) * (B * H) + (size_t)b * H + u0 + u] = (_Float16)h;
      if (out16) out16[row * H + u0 + u] = (_Float16)h;
      if (out32) out32[row * H + u0 + u] = h;
    }

    __threadfence();
    __syncthreads();
    if (tid == 0)
      __hip_atomic_fetch_add(bar + t, 1, __ATOMIC_RELEASE, __HIP_MEMORY_SCOPE_AGENT);
  }
}

// ---------------- host launcher ----------------

extern "C" void kernel_launch(void* const* d_in, const int* in_sizes, int n_in,
                              void* d_out, int out_size, void* d_ws, size_t ws_size,
                              hipStream_t stream) {
  const float* x     = (const float*)d_in[0];
  const float* W_ih0 = (const float*)d_in[1];
  const float* W_hh0 = (const float*)d_in[2];
  const float* b_ih0 = (const float*)d_in[3];
  const float* b_hh0 = (const float*)d_in[4];
  const float* W_ih1 = (const float*)d_in[5];
  const float* W_hh1 = (const float*)d_in[6];
  const float* b_ih1 = (const float*)d_in[7];
  const float* b_hh1 = (const float*)d_in[8];
  float* out = (float*)d_out;

  char* ws = (char*)d_ws;
  int* bar0 = (int*)ws;                         // [T]
  int* bar1 = bar0 + T;                         // [T]
  size_t pos = 65536;
  _Float16* x16 = (_Float16*)(ws + pos);  pos += (size_t)BT * 1024 * 2;  // 64 MB
  _Float16* w16 = (_Float16*)(ws + pos);  pos += (size_t)G * 1024 * 2;   // 8 MB
  _Float16* whh16 = (_Float16*)(ws + pos); pos += (size_t)G * 1024 * 2;  // 8 MB
  float* bsum = (float*)(ws + pos);       pos += 65536;
  _Float16* hbuf = (_Float16*)(ws + pos); pos += (size_t)2 * B * H * 2;  // 256 KB
  _Float16* xg = (_Float16*)(ws + pos);   // [BT, G] f16 : 256 MB

  const int TPB = 256;
  auto nblk = [](long n) { return (unsigned)((n + 255) / 256); };

  zero_u32_kernel<<<64, TPB, 0, stream>>>((uint32_t*)bar0, 2L * T);
  zero_u32_kernel<<<64, TPB, 0, stream>>>((uint32_t*)hbuf, (long)2 * B * H / 2);

  // ---------------- layer 0 ----------------
  cvt_pad_f16_kernel<<<nblk((long)BT * DINP), TPB, 0, stream>>>(x, x16, BT, DIN, DINP);
  cvt_pad_f16_kernel<<<nblk((long)G * DINP), TPB, 0, stream>>>(W_ih0, w16, G, DIN, DINP);
  cvt_pad_f16_kernel<<<nblk((long)G * H), TPB, 0, stream>>>(W_hh0, whh16, G, H, H);
  bias_sum_kernel<<<G / TPB, TPB, 0, stream>>>(b_ih0, b_hh0, bsum, G);

  wmma_gemm_bias_kernel<<<dim3(G / 64, BT / 64), TPB, 0, stream>>>(
      x16, w16, bsum, xg, BT, G, DINP);

  lstm_rec_kernel<<<REC_BLOCKS, TPB, SMEM_REC, stream>>>(
      xg, whh16, hbuf, bar0, /*out16=*/x16, /*out32=*/nullptr);

  // ---------------- layer 1 ----------------
  zero_u32_kernel<<<64, TPB, 0, stream>>>((uint32_t*)hbuf, (long)B * H / 2);
  cvt_pad_f16_kernel<<<nblk((long)G * H), TPB, 0, stream>>>(W_ih1, w16, G, H, H);
  cvt_pad_f16_kernel<<<nblk((long)G * H), TPB, 0, stream>>>(W_hh1, whh16, G, H, H);
  bias_sum_kernel<<<G / TPB, TPB, 0, stream>>>(b_ih1, b_hh1, bsum, G);

  wmma_gemm_bias_kernel<<<dim3(G / 64, BT / 64), TPB, 0, stream>>>(
      x16, w16, bsum, xg, BT, G, H);

  lstm_rec_kernel<<<REC_BLOCKS, TPB, SMEM_REC, stream>>>(
      xg, whh16, hbuf, bar1, /*out16=*/nullptr, /*out32=*/out);
}
